// GaussiansGenerator_38001870635132
// MI455X (gfx1250) — matrix operand order repaired and checked by reading, not compiled, not run on hardware
//
#include <hip/hip_runtime.h>
#include <stdint.h>

// ---------------- problem constants ----------------
#define B_   4
#define N_   4096
#define K_   20

// ---------------- BN stat buffer layout (float offsets in STATS) -----------
#define SUM1 0
#define SSQ1 64
#define SUMX 128
#define SSQX 256
#define SUM2 384
#define SSQ2 512
#define M1O  640
#define I1O  704
#define MXO  768
#define IXO  896
#define M2O  1024
#define I2O  1152
#define STATS_N 1280

// ---------------- WMMA helpers (gfx1250, wave32) ----------------
typedef __attribute__((ext_vector_type(16))) _Float16 v16h;
typedef __attribute__((ext_vector_type(8)))  float    v8f;

union V16H { v16h v; uint32_t u[8]; };

// A fragment: 16x32 f16 (M x K). Rows = 16 query points stored as 64 dwords
// (128 packed f16) each in LDS. ISA layout: lane<16 -> K in {0..7,16..23},
// lane>=16 -> K in {8..15,24..31} (plus k0 window offset).
static __device__ __forceinline__ v16h load_A(const uint32_t* q, int k0, int lane) {
  int half = lane >> 4, m = lane & 15;
  const uint32_t* row = q + m * 64;
  V16H a;
#pragma unroll
  for (int r = 0; r < 4; ++r) {
    a.u[r]     = row[(k0 + 8 * half + 2 * r) >> 1];
    a.u[r + 4] = row[(k0 + 16 + 8 * half + 2 * r) >> 1];
  }
  return a.v;
}

// B fragment: 32x16 f16 (K x N). Columns = 16 candidate points (64 dwords each).
// lane<16 -> N=lane, K in {0..7,16..23}; lane>=16 -> N=lane-16, K in {8..15,24..31} (+k0).
static __device__ __forceinline__ v16h load_B(const uint32_t* c, int k0, int lane) {
  int half = lane >> 4, n = lane & 15;
  const uint32_t* col = c + n * 64;
  V16H b;
#pragma unroll
  for (int r = 0; r < 8; ++r)
    b.u[r] = col[(k0 + 8 * half + 2 * r + (r >= 4 ? 8 : 0)) >> 1];
  return b.v;
}

// ---------------- kernels ----------------

// style MLP: zn = z/(|z|+eps); in = [x(3), zn(128)]; y1=lrelu(W1 in+b1,.01);
// y2=lrelu(W2 y1+b2,.01) -> style (B,128,N)
__global__ __launch_bounds__(128) void k_style(
    const float* __restrict__ x, const float* __restrict__ z,
    const float* __restrict__ w1, const float* __restrict__ b1,
    const float* __restrict__ w2, const float* __restrict__ b2,
    float* __restrict__ style) {
  __shared__ float sin_[131];
  __shared__ float sy[128];
  __shared__ float red[128];
  int t = threadIdx.x;
  int p = blockIdx.x;
  int b = p / N_, n = p % N_;
  float zv = z[(size_t)p * 128 + t];
  red[t] = zv * zv;
  __syncthreads();
  for (int s = 64; s > 0; s >>= 1) {
    if (t < s) red[t] += red[t + s];
    __syncthreads();
  }
  float nrm = sqrtf(red[0]) + 1e-8f;
  sin_[3 + t] = zv / nrm;
  if (t < 3) sin_[t] = x[(size_t)p * 3 + t];
  __syncthreads();
  float a = b1[t];
  for (int i = 0; i < 131; ++i) a += w1[t * 131 + i] * sin_[i];
  a = a >= 0.f ? a : 0.01f * a;
  sy[t] = a;
  __syncthreads();
  float c = b2[t];
  for (int i = 0; i < 128; ++i) c += w2[t * 128 + i] * sy[i];
  c = c >= 0.f ? c : 0.01f * c;
  style[((size_t)b * 128 + t) * N_ + n] = c;
}

// pack features to f16 (B,N,128) zero-padded, squared norms, optional f32
// channel-major transpose copy (for block-0 whose input arrives point-major).
__global__ __launch_bounds__(256) void k_prep(
    const float* __restrict__ src, unsigned short* __restrict__ xh,
    float* __restrict__ sqn, float* __restrict__ xt, int fin, int layoutCN) {
  int p = blockIdx.x * 256 + threadIdx.x;
  if (p >= B_ * N_) return;
  int b = p / N_, n = p % N_;
  unsigned short* dst = xh + (size_t)p * 128;
  float s = 0.f;
  for (int c = 0; c < 128; ++c) {
    float v = 0.f;
    if (c < fin)
      v = layoutCN ? src[((size_t)b * fin + c) * N_ + n] : src[(size_t)p * fin + c];
    union { _Float16 h; unsigned short u; } cv;
    cv.h = (_Float16)v;
    dst[c] = cv.u;
    s += v * v;
    if (xt && c < fin) xt[((size_t)b * fin + c) * N_ + n] = v;
  }
  sqn[p] = s;
}

// kNN via WMMA gram tiles. 4 waves/WG; each wave owns 16 query rows;
// candidates streamed in 16-wide tiles through shared LDS.
__global__ __launch_bounds__(128) void k_knn(
    const unsigned short* __restrict__ xh, const float* __restrict__ sqn,
    int* __restrict__ idxbuf) {
  __shared__ uint32_t sQ[64 * 64];   // 64 queries x 128 f16
  __shared__ uint32_t sC[16 * 64];   // 16 candidates x 128 f16
  __shared__ float    sD[4][256];    // per-wave 16x16 distance tile
  __shared__ float    sQn[64];
  __shared__ float    sCn[16];
  int tid = threadIdx.x, wave = tid >> 5, lane = tid & 31;
  int b = blockIdx.y, qbase = blockIdx.x * 64;
  const uint32_t* xb = (const uint32_t*)xh + (size_t)b * N_ * 64;
  const float* qn = sqn + (size_t)b * N_;
  for (int i = tid; i < 64 * 64; i += 128) sQ[i] = xb[(size_t)qbase * 64 + i];
  if (tid < 64) sQn[tid] = qn[qbase + tid];
  __syncthreads();

  float bd[21];
  int   bi[21];
#pragma unroll
  for (int j = 0; j < 21; ++j) { bd[j] = 3.4e38f; bi[j] = -1; }

  const uint32_t* qrow = &sQ[wave * 16 * 64];
  for (int ct = 0; ct < N_ / 16; ++ct) {
    __syncthreads();
    for (int i = tid; i < 16 * 64; i += 128) sC[i] = xb[(size_t)ct * 1024 + i];
    if (tid < 16) sCn[tid] = qn[ct * 16 + tid];
    if (ct + 1 < N_ / 16)  // prefetch next candidate tile (global_prefetch_b8)
      __builtin_prefetch((const void*)(xb + (size_t)(ct + 1) * 1024 + tid * 8), 0, 1);
    __syncthreads();

    v8f acc = {};
#pragma unroll
    for (int k0 = 0; k0 < 128; k0 += 32) {
      v16h a  = load_A(qrow, k0, lane);
      v16h bm = load_B(sC, k0, lane);
      acc = __builtin_amdgcn_wmma_f32_16x16x32_f16(
          false, a, false, bm, (short)0, acc, false, false);
    }
    {
      int n = lane & 15, mh = (lane >> 4) * 8;
#pragma unroll
      for (int r = 0; r < 8; ++r)
        sD[wave][(mh + r) * 16 + n] =
            sQn[wave * 16 + mh + r] + sCn[n] - 2.0f * acc[r];
    }
    if (lane < 16) {
      int m = lane;
      for (int n2 = 0; n2 < 16; ++n2) {
        float d = sD[wave][m * 16 + n2];
        if (d < bd[20]) {
          int ci = ct * 16 + n2;
          int j = 20;
          while (j > 0 && bd[j - 1] > d) {
            bd[j] = bd[j - 1]; bi[j] = bi[j - 1]; --j;
          }
          bd[j] = d; bi[j] = ci;
        }
      }
    }
  }
  if (lane < 16) {
    int q = qbase + wave * 16 + lane;
    for (int j = 1; j < 21; ++j)
      idxbuf[((size_t)b * N_ + q) * K_ + (j - 1)] = bi[j];
  }
}

__global__ void k_zero(float* p, int nfl) {
  for (int i = threadIdx.x; i < nfl; i += blockDim.x) p[i] = 0.f;
}

__global__ void k_finbn(float* stats, int sumO, int ssqO, int mO, int iO,
                        int nch, float cnt) {
  int t = threadIdx.x;
  if (t < nch) {
    float m = stats[sumO + t] / cnt;
    float v = stats[ssqO + t] / cnt - m * m;
    stats[mO + t] = m;
    stats[iO + t] = rsqrtf(v + 1e-5f);
  }
}

// BN stats pass 1: t1 = W1*(neigh-central) (64ch) and tx = Wx*ee (128ch).
// Conv biases cancel under BN and are omitted everywhere pre-BN.
__global__ __launch_bounds__(128) void k_stat1(
    const float* __restrict__ feat, const int* __restrict__ idx,
    const float* __restrict__ w1, const float* __restrict__ wx,
    float* __restrict__ stats, int fin) {
  __shared__ float sCen[128];
  __shared__ float sD[128];
  const int PTS = 8;
  int t = threadIdx.x;
  int gp = blockIdx.x * PTS;
  float s1 = 0.f, ss1 = 0.f, sx = 0.f, ssx = 0.f;
  for (int p = 0; p < PTS; ++p) {
    int point = gp + p;
    int b = point / N_, n = point % N_;
    const float* fb = feat + (size_t)b * fin * N_;
    const int* irow = idx + ((size_t)b * N_ + n) * K_;
    __syncthreads();
    for (int i = t; i < fin; i += 128) sCen[i] = fb[(size_t)i * N_ + n];
    __syncthreads();
    float txc = 0.f;
    for (int i = 0; i < fin; ++i) txc += wx[t * 2 * fin + i] * sCen[i];
    for (int k = 0; k < K_; ++k) {
      int nj = irow[k];
      __syncthreads();
      for (int i = t; i < fin; i += 128) sD[i] = fb[(size_t)i * N_ + nj] - sCen[i];
      __syncthreads();
      if (t < 64) {
        float v = 0.f;
        for (int i = 0; i < fin; ++i) v += w1[t * fin + i] * sD[i];
        s1 += v; ss1 += v * v;
      }
      float vx = txc;
      for (int i = 0; i < fin; ++i) vx += wx[t * 2 * fin + fin + i] * sD[i];
      sx += vx; ssx += vx * vx;
    }
  }
  if (t < 64) {
    atomicAdd(&stats[SUM1 + t], s1);
    atomicAdd(&stats[SSQ1 + t], ss1);
  }
  atomicAdd(&stats[SUMX + t], sx);
  atomicAdd(&stats[SSQX + t], ssx);
}

// BN stats pass 2: t2 = W2*lrelu(bn1(t1)).
__global__ __launch_bounds__(128) void k_stat2(
    const float* __restrict__ feat, const int* __restrict__ idx,
    const float* __restrict__ w1, const float* __restrict__ w2,
    float* __restrict__ stats, const float* __restrict__ g1,
    const float* __restrict__ bt1, int fin) {
  __shared__ float sCen[128];
  __shared__ float sD[128];
  __shared__ float sU[64];
  const int PTS = 8;
  int t = threadIdx.x;
  int gp = blockIdx.x * PTS;
  float m1_ = 0.f, i1_ = 0.f, g1_ = 0.f, b1_ = 0.f;
  if (t < 64) {
    m1_ = stats[M1O + t]; i1_ = stats[I1O + t];
    g1_ = g1[t]; b1_ = bt1[t];
  }
  float s2 = 0.f, ss2 = 0.f;
  for (int p = 0; p < PTS; ++p) {
    int point = gp + p;
    int b = point / N_, n = point % N_;
    const float* fb = feat + (size_t)b * fin * N_;
    const int* irow = idx + ((size_t)b * N_ + n) * K_;
    __syncthreads();
    for (int i = t; i < fin; i += 128) sCen[i] = fb[(size_t)i * N_ + n];
    __syncthreads();
    for (int k = 0; k < K_; ++k) {
      int nj = irow[k];
      __syncthreads();
      for (int i = t; i < fin; i += 128) sD[i] = fb[(size_t)i * N_ + nj] - sCen[i];
      __syncthreads();
      if (t < 64) {
        float v = 0.f;
        for (int i = 0; i < fin; ++i) v += w1[t * fin + i] * sD[i];
        float u = (v - m1_) * i1_ * g1_ + b1_;
        sU[t] = u >= 0.f ? u : 0.2f * u;
      }
      __syncthreads();
      float v2 = 0.f;
      for (int i = 0; i < 64; ++i) v2 += w2[t * 64 + i] * sU[i];
      s2 += v2; ss2 += v2 * v2;
    }
  }
  atomicAdd(&stats[SUM2 + t], s2);
  atomicAdd(&stats[SSQ2 + t], ss2);
}

// swizzle wo (128 x 128 x 20 f32) into f16 A-fragment lane order:
// [mtile(8)][ktile(80)][lane(32)][8 dwords], inner dim K = i*20+k (2560).
__global__ __launch_bounds__(256) void k_swizzle_wo(
    const float* __restrict__ wo, unsigned int* __restrict__ wsw) {
  int tid = blockIdx.x * 256 + threadIdx.x;
  if (tid >= 8 * 80 * 32) return;
  int lane = tid & 31;
  int kt = (tid >> 5) % 80;
  int mtile = tid / (80 * 32);
  int half = lane >> 4, m = lane & 15;
  int o = mtile * 16 + m;
  const float* row = wo + (size_t)o * 2560;
  unsigned int* dst = wsw + (size_t)tid * 8;
#pragma unroll
  for (int r = 0; r < 8; ++r) {
    int kk = (r < 4) ? (kt * 32 + 8 * half + 2 * r)
                     : (kt * 32 + 16 + 8 * half + 2 * (r - 4));
    union { _Float16 h[2]; unsigned int u; } cv;
    cv.h[0] = (_Float16)row[kk];
    cv.h[1] = (_Float16)row[kk + 1];
    dst[r] = cv.u;
  }
}

// fused edge block, 16 points per WG.
// Phase 1 (per point): t1->bn1->lrelu, t2->bn2->lrelu, softmax(k),
//   tx->bnx->lrelu, h = hx*w, packed f16 into LDS B-matrix (K=2560 x N=16).
// Phase 2: out[o,p] = sum_K woF16[o,K]*h[K,p] via v_wmma_f32_16x16x32_f16,
//   8 M-tiles x 80 K-steps, 2 M-tiles per wave, + bo.
__global__ __launch_bounds__(128) void k_pointC(
    const float* __restrict__ feat, const int* __restrict__ idx,
    const float* __restrict__ w1, const float* __restrict__ w2,
    const float* __restrict__ wx, const unsigned int* __restrict__ wo_sw,
    const float* __restrict__ bo, const float* __restrict__ stats,
    const float* __restrict__ g1, const float* __restrict__ bt1,
    const float* __restrict__ g2, const float* __restrict__ bt2,
    const float* __restrict__ gx, const float* __restrict__ btx,
    float* __restrict__ pre, int fin) {
  extern __shared__ char smem[];
  float*    sCen = (float*)smem;              // 128
  float*    sD   = sCen + 128;                // 128
  float*    sU   = sD + 128;                  // 64
  float*    sL   = sU + 64;                   // 128*20
  float*    sHx  = sL + 128 * K_;             // 128*20
  uint32_t* sHf  = (uint32_t*)(sHx + 128 * K_);  // 16*1280 dwords (80KB)

  int t = threadIdx.x;
  int lane = t & 31, wave = t >> 5;
  int pbase = blockIdx.x * 16;                // 16 consecutive points, one batch
  int b = pbase / N_;
  const float* fb = feat + (size_t)b * fin * N_;

  float m1_ = 0.f, i1_ = 0.f, g1_ = 0.f, b1_ = 0.f;
  if (t < 64) {
    m1_ = stats[M1O + t]; i1_ = stats[I1O + t];
    g1_ = g1[t]; b1_ = bt1[t];
  }
  float m2_ = stats[M2O + t], i2_ = stats[I2O + t], g2_ = g2[t], b2_ = bt2[t];
  float mx_ = stats[MXO + t], ix_ = stats[IXO + t], gx_ = gx[t], bx_ = btx[t];

  for (int p = 0; p < 16; ++p) {
    int n = (pbase + p) % N_;
    const int* irow = idx + ((size_t)b * N_ + n) * K_;
    __syncthreads();
    for (int i = t; i < fin; i += 128) sCen[i] = fb[(size_t)i * N_ + n];
    __syncthreads();
    float txc = 0.f;
    for (int i = 0; i < fin; ++i) txc += wx[t * 2 * fin + i] * sCen[i];

    for (int k = 0; k < K_; ++k) {
      int nj = irow[k];
      __syncthreads();
      for (int i = t; i < fin; i += 128) sD[i] = fb[(size_t)i * N_ + nj] - sCen[i];
      __syncthreads();
      if (t < 64) {
        float v = 0.f;
        for (int i = 0; i < fin; ++i) v += w1[t * fin + i] * sD[i];
        float u = (v - m1_) * i1_ * g1_ + b1_;
        sU[t] = u >= 0.f ? u : 0.2f * u;
      }
      __syncthreads();
      float v2 = 0.f;
      for (int i = 0; i < 64; ++i) v2 += w2[t * 64 + i] * sU[i];
      float l = (v2 - m2_) * i2_ * g2_ + b2_;
      sL[t * K_ + k] = l >= 0.f ? l : 0.2f * l;
      float vx = txc;
      for (int i = 0; i < fin; ++i) vx += wx[t * 2 * fin + fin + i] * sD[i];
      float hxx = (vx - mx_) * ix_ * gx_ + bx_;
      sHx[t * K_ + k] = hxx >= 0.f ? hxx : 0.2f * hxx;
    }
    // softmax over k (own row), multiply, pack f16 pairs into column p
    float mxv = -3.4e38f;
    for (int k = 0; k < K_; ++k) mxv = fmaxf(mxv, sL[t * K_ + k]);
    float se = 0.f;
    for (int k = 0; k < K_; ++k) {
      float ev = __expf(sL[t * K_ + k] - mxv);
      sL[t * K_ + k] = ev;
      se += ev;
    }
    float inv = 1.f / se;
    uint32_t* colw = sHf + p * 1280 + t * 10;   // K = t*20 .. t*20+19
    for (int j = 0; j < 10; ++j) {
      float h0 = sHx[t * K_ + 2 * j] * sL[t * K_ + 2 * j] * inv;
      float h1 = sHx[t * K_ + 2 * j + 1] * sL[t * K_ + 2 * j + 1] * inv;
      union { _Float16 h[2]; uint32_t u; } cv;
      cv.h[0] = (_Float16)h0;
      cv.h[1] = (_Float16)h1;
      colw[j] = cv.u;
    }
  }
  __syncthreads();

  // ---- Phase 2: WMMA wo contraction: D(128x16) = wo(128x2560) * H(2560x16)
  int half = lane >> 4, p16 = lane & 15;
  const uint32_t* colb = sHf + p16 * 1280 + 4 * half;
  int ncol = (pbase + p16) % N_;
  for (int mt = wave; mt < 8; mt += 4) {
    v8f acc = {};
    for (int kt = 0; kt < 80; ++kt) {
      V16H a;
      const uint4* ap =
          (const uint4*)(wo_sw + ((((size_t)mt * 80 + kt) * 32 + lane) * 8));
      uint4 x0 = ap[0], x1 = ap[1];
      a.u[0] = x0.x; a.u[1] = x0.y; a.u[2] = x0.z; a.u[3] = x0.w;
      a.u[4] = x1.x; a.u[5] = x1.y; a.u[6] = x1.z; a.u[7] = x1.w;
      V16H bm;
      const uint4* bp4 = (const uint4*)(colb + kt * 16);
      uint4 y0 = bp4[0], y1 = bp4[2];
      bm.u[0] = y0.x; bm.u[1] = y0.y; bm.u[2] = y0.z; bm.u[3] = y0.w;
      bm.u[4] = y1.x; bm.u[5] = y1.y; bm.u[6] = y1.z; bm.u[7] = y1.w;
      acc = __builtin_amdgcn_wmma_f32_16x16x32_f16(
          false, a.v, false, bm.v, (short)0, acc, false, false);
    }
#pragma unroll
    for (int r = 0; r < 8; ++r) {
      int o = mt * 16 + r + 8 * half;
      float vout = acc[r] + bo[o];
      pre[((size_t)b * 128 + o) * N_ + ncol] = vout;
    }
  }
}

// instance-norm stats over points for each (b,c)
__global__ __launch_bounds__(256) void k_inststat(
    const float* __restrict__ pre, float* __restrict__ inst) {
  int bc = blockIdx.x;
  __shared__ float rs[256], rq[256];
  const float* row = pre + (size_t)bc * N_;
  float s = 0.f, q = 0.f;
  for (int n = threadIdx.x; n < N_; n += 256) {
    float v = row[n];
    s += v; q += v * v;
  }
  rs[threadIdx.x] = s; rq[threadIdx.x] = q;
  __syncthreads();
  for (int st = 128; st > 0; st >>= 1) {
    if (threadIdx.x < st) {
      rs[threadIdx.x] += rs[threadIdx.x + st];
      rq[threadIdx.x] += rq[threadIdx.x + st];
    }
    __syncthreads();
  }
  if (threadIdx.x == 0) {
    float m = rs[0] / (float)N_;
    float v = rq[0] / (float)N_ - m * m;
    inst[bc * 2] = m;
    inst[bc * 2 + 1] = rsqrtf(v + 1e-5f);
  }
}

// AdaIN: s = Ws*style + bs per point; gamma*(x-m)*istd + beta; lrelu 0.2
__global__ __launch_bounds__(128) void k_adain(
    const float* __restrict__ pre, const float* __restrict__ style,
    const float* __restrict__ wsM, const float* __restrict__ bsV,
    const float* __restrict__ inst, float* __restrict__ hout) {
  __shared__ float sSty[128];
  int t = threadIdx.x;
  int point = blockIdx.x;
  int b = point / N_, n = point % N_;
  sSty[t] = style[((size_t)b * 128 + t) * N_ + n];
  __syncthreads();
  float gm = bsV[t], bt = bsV[128 + t];
  for (int i = 0; i < 128; ++i) {
    gm += wsM[t * 128 + i] * sSty[i];
    bt += wsM[(128 + t) * 128 + i] * sSty[i];
  }
  float m = inst[(b * 128 + t) * 2], is = inst[(b * 128 + t) * 2 + 1];
  float v = pre[((size_t)b * 128 + t) * N_ + n];
  float o = gm * ((v - m) * is) + bt;
  hout[((size_t)b * 128 + t) * N_ + n] = o >= 0.f ? o : 0.2f * o;
}

__global__ __launch_bounds__(256) void k_maxpool(
    const float* __restrict__ h, float* __restrict__ fg) {
  int bc = blockIdx.x;
  __shared__ float r[256];
  const float* row = h + (size_t)bc * N_;
  float m = -3.4e38f;
  for (int n = threadIdx.x; n < N_; n += 256) m = fmaxf(m, row[n]);
  r[threadIdx.x] = m;
  __syncthreads();
  for (int s = 128; s > 0; s >>= 1) {
    if (threadIdx.x < s) r[threadIdx.x] = fmaxf(r[threadIdx.x], r[threadIdx.x + s]);
    __syncthreads();
  }
  if (threadIdx.x == 0) fg[bc] = r[0];
}

// global head: two FCs with batch(B=4) BN + lrelu 0.01 in one WG
__global__ __launch_bounds__(512) void k_head(
    const float* __restrict__ fgmax,
    const float* __restrict__ g1w, const float* __restrict__ g1b,
    const float* __restrict__ g1g, const float* __restrict__ g1bt,
    const float* __restrict__ g2w, const float* __restrict__ g2b,
    const float* __restrict__ g2g, const float* __restrict__ g2bt,
    float* __restrict__ fg2) {
  __shared__ float sfg[512];
  __shared__ float sy1[512];
  int t = threadIdx.x;
  sfg[t] = fgmax[t];
  __syncthreads();
  if (t < 128) {
    float y[4];
    for (int b = 0; b < 4; ++b) {
      float a = g1b[t];
      for (int i = 0; i < 128; ++i) a += g1w[t * 128 + i] * sfg[b * 128 + i];
      y[b] = a;
    }
    float m = 0.25f * (y[0] + y[1] + y[2] + y[3]);
    float v = 0.f;
    for (int b = 0; b < 4; ++b) { float d = y[b] - m; v += d * d; }
    v *= 0.25f;
    float is = rsqrtf(v + 1e-5f);
    for (int b = 0; b < 4; ++b) {
      float o = (y[b] - m) * is * g1g[t] + g1bt[t];
      sy1[b * 128 + t] = o >= 0.f ? o : 0.01f * o;
    }
  }
  __syncthreads();
  {
    float y[4];
    for (int b = 0; b < 4; ++b) {
      float a = g2b[t];
      for (int i = 0; i < 128; ++i) a += g2w[t * 128 + i] * sy1[b * 128 + i];
      y[b] = a;
    }
    float m = 0.25f * (y[0] + y[1] + y[2] + y[3]);
    float v = 0.f;
    for (int b = 0; b < 4; ++b) { float d = y[b] - m; v += d * d; }
    v *= 0.25f;
    float is = rsqrtf(v + 1e-5f);
    for (int b = 0; b < 4; ++b) {
      float o = (y[b] - m) * is * g2g[t] + g2bt[t];
      fg2[b * 512 + t] = o >= 0.f ? o : 0.01f * o;
    }
  }
}

// out (B,N,640): [fg2 broadcast (512) | h2 (128)]
__global__ __launch_bounds__(256) void k_assemble(
    const float* __restrict__ fg2, const float* __restrict__ h,
    float* __restrict__ out) {
  size_t e = (size_t)blockIdx.x * 256 + threadIdx.x;
  int c = (int)(e % 640);
  size_t bn = e / 640;
  int n = (int)(bn % N_);
  int b = (int)(bn / N_);
  out[e] = (c < 512) ? fg2[b * 512 + c]
                     : h[((size_t)b * 128 + (c - 512)) * N_ + n];
}

// ---------------- launcher ----------------
extern "C" void kernel_launch(void* const* d_in, const int* in_sizes, int n_in,
                              void* d_out, int out_size, void* d_ws, size_t ws_size,
                              hipStream_t stream) {
  (void)in_sizes; (void)n_in; (void)out_size; (void)ws_size;
  const float* x   = (const float*)d_in[0];
  const float* z   = (const float*)d_in[1];
  const float* h1w = (const float*)d_in[2];
  const float* h1b = (const float*)d_in[3];
  const float* h2w = (const float*)d_in[4];
  const float* h2b = (const float*)d_in[5];
  const float* g1w = (const float*)d_in[6];
  const float* g1b = (const float*)d_in[7];
  const float* g1g = (const float*)d_in[8];
  const float* g1bt= (const float*)d_in[9];
  const float* g2w = (const float*)d_in[10];
  const float* g2b = (const float*)d_in[11];
  const float* g2g = (const float*)d_in[12];
  const float* g2bt= (const float*)d_in[13];

  // workspace carve
  char* w = (char*)d_ws;
  size_t off = 0;
  auto alloc = [&](size_t bytes) -> void* {
    void* p = w + off;
    off = (off + bytes + 255) & ~(size_t)255;
    return p;
  };
  float*          STYLE = (float*)alloc((size_t)B_ * 128 * N_ * 4);
  unsigned short* XH    = (unsigned short*)alloc((size_t)B_ * N_ * 128 * 2);
  float*          SQN   = (float*)alloc((size_t)B_ * N_ * 4);
  int*            IDX   = (int*)alloc((size_t)B_ * N_ * K_ * 4);
  float*          XT    = (float*)alloc((size_t)B_ * 3 * N_ * 4);
  float*          HA    = (float*)alloc((size_t)B_ * 128 * N_ * 4);
  float*          HB    = (float*)alloc((size_t)B_ * 128 * N_ * 4);
  float*          PRE   = (float*)alloc((size_t)B_ * 128 * N_ * 4);
  float*          STATS = (float*)alloc((size_t)STATS_N * 4);
  float*          INST  = (float*)alloc((size_t)B_ * 128 * 2 * 4);
  float*          FGMAX = (float*)alloc((size_t)B_ * 128 * 4);
  float*          FG2   = (float*)alloc((size_t)B_ * 512 * 4);
  unsigned int*   WOSW  = (unsigned int*)alloc((size_t)8 * 80 * 32 * 8 * 4);

  k_style<<<B_ * N_, 128, 0, stream>>>(x, z, h1w, h1b, h2w, h2b, STYLE);

  const float cnt = (float)((size_t)B_ * N_ * K_);
  const size_t SMEM_PC = (size_t)(128 + 128 + 64 + 128 * K_ + 128 * K_) * 4 +
                         (size_t)16 * 1280 * 4;  // ~101.3 KB
  for (int blk = 0; blk < 2; ++blk) {
    int base = 14 + blk * 16;
    const float* w1   = (const float*)d_in[base + 0];
    const float* bn1g = (const float*)d_in[base + 2];
    const float* bn1b = (const float*)d_in[base + 3];
    const float* w2   = (const float*)d_in[base + 4];
    const float* bn2g = (const float*)d_in[base + 6];
    const float* bn2b = (const float*)d_in[base + 7];
    const float* wx   = (const float*)d_in[base + 8];
    const float* bnxg = (const float*)d_in[base + 10];
    const float* bnxb = (const float*)d_in[base + 11];
    const float* wo   = (const float*)d_in[base + 12];
    const float* bo   = (const float*)d_in[base + 13];
    const float* wsm  = (const float*)d_in[base + 14];
    const float* bsv  = (const float*)d_in[base + 15];
    int fin = (blk == 0) ? 3 : 128;
    const float* feat;
    float* hout;
    if (blk == 0) {
      k_prep<<<(B_ * N_) / 256, 256, 0, stream>>>(x, XH, SQN, XT, 3, 0);
      feat = XT; hout = HA;
    } else {
      k_prep<<<(B_ * N_) / 256, 256, 0, stream>>>(HA, XH, SQN, nullptr, 128, 1);
      feat = HA; hout = HB;
    }
    k_knn<<<dim3(N_ / 64, B_), 128, 0, stream>>>(XH, SQN, IDX);
    k_zero<<<1, 256, 0, stream>>>(STATS, STATS_N);
    k_stat1<<<(B_ * N_) / 8, 128, 0, stream>>>(feat, IDX, w1, wx, STATS, fin);
    k_finbn<<<1, 256, 0, stream>>>(STATS, SUM1, SSQ1, M1O, I1O, 64, cnt);
    k_finbn<<<1, 256, 0, stream>>>(STATS, SUMX, SSQX, MXO, IXO, 128, cnt);
    k_stat2<<<(B_ * N_) / 8, 128, 0, stream>>>(feat, IDX, w1, w2, STATS, bn1g, bn1b, fin);
    k_finbn<<<1, 256, 0, stream>>>(STATS, SUM2, SSQ2, M2O, I2O, 128, cnt);
    k_swizzle_wo<<<80, 256, 0, stream>>>(wo, WOSW);
    k_pointC<<<(B_ * N_) / 16, 128, SMEM_PC, stream>>>(
        feat, IDX, w1, w2, wx, WOSW, bo, STATS,
        bn1g, bn1b, bn2g, bn2b, bnxg, bnxb, PRE, fin);
    k_inststat<<<B_ * 128, 256, 0, stream>>>(PRE, INST);
    k_adain<<<B_ * N_, 128, 0, stream>>>(PRE, STYLE, wsm, bsv, INST, hout);
  }

  k_maxpool<<<B_ * 128, 256, 0, stream>>>(HB, FGMAX);
  k_head<<<1, 512, 0, stream>>>(FGMAX, g1w, g1b, g1g, g1bt, g2w, g2b, g2g, g2bt, FG2);
  k_assemble<<<(B_ * N_ * 640) / 256, 256, 0, stream>>>(FG2, HB, (float*)d_out);
}
